// Attention_10230612099272
// MI455X (gfx1250) — matrix operand order repaired and checked by reading, not compiled, not run on hardware
//
#include <hip/hip_runtime.h>
#include <hip/hip_bf16.h>
#include <math.h>

typedef __attribute__((ext_vector_type(16))) __bf16 v16bf;
typedef __attribute__((ext_vector_type(8)))  float  v8f;

union FragBF { v16bf v; unsigned u[8]; };

__device__ __forceinline__ v8f wmma_bf16(v16bf a, v16bf b, v8f c) {
  // D = A(16x32 bf16) * B(32x16 bf16) + C(16x16 f32)
  return __builtin_amdgcn_wmma_f32_16x16x32_bf16(false, a, false, b, (short)0, c, false, false);
}

// ---- CDNA5 async copy: 16B global -> LDS, tracked by ASYNCcnt ----
__device__ __forceinline__ void async_copy_b128(unsigned lds_byte_off, const void* gaddr) {
  asm volatile("global_load_async_to_lds_b128 %0, %1, off"
               :: "v"(lds_byte_off), "v"(gaddr) : "memory");
}
template<int N>
__device__ __forceinline__ void wait_asynccnt() {
  asm volatile("s_wait_asynccnt %0" :: "i"(N) : "memory");
}
__device__ __forceinline__ unsigned lds_off(const void* p) {
  return (unsigned)(size_t)p;   // low 32 bits of generic pointer = LDS offset
}

// ---------------------------------------------------------------------------
// Prep: f32 -> bf16 copy (row-major preserved)
// ---------------------------------------------------------------------------
__global__ void to_bf16_kernel(const float* __restrict__ src, __bf16* __restrict__ dst,
                               size_t n) {
  size_t i = ((size_t)blockIdx.x * blockDim.x + threadIdx.x) * 4;
  if (i + 3 < n) {
    float4 f = *(const float4*)(src + i);
    dst[i]     = (__bf16)f.x;
    dst[i + 1] = (__bf16)f.y;
    dst[i + 2] = (__bf16)f.z;
    dst[i + 3] = (__bf16)f.w;
  }
}

// Prep: W[K,N] f32 -> WT[N,K] bf16 (LDS-tiled transpose, coalesced both sides)
__global__ __launch_bounds__(256) void transpose_bf16_kernel(
    const float* __restrict__ W, __bf16* __restrict__ WT, int K, int N) {
  __shared__ __bf16 tile[32][33];
  const int bx = blockIdx.x * 32;           // n
  const int by = blockIdx.y * 32;           // k
  const int tx = threadIdx.x & 31, ty = threadIdx.x >> 5;
#pragma unroll
  for (int i = 0; i < 32; i += 8)
    tile[ty + i][tx] = (__bf16)W[(size_t)(by + ty + i) * N + bx + tx];
  __syncthreads();
#pragma unroll
  for (int i = 0; i < 32; i += 8)
    WT[(size_t)(bx + ty + i) * K + by + tx] = tile[tx][ty + i];
}

// ---------------------------------------------------------------------------
// GEMM: C[M,N] = A[M,K] * W[K,N] + bias, all operands bf16, f32 accumulate.
// A:  AMODE 0: bf16 row-major [M,K]
//     AMODE 1: bf16 head-major [B,NH,S,HD] (row=b*S+s, col=h*HD+d; 32-chunks
//              stay inside one head since HD%32==0)
// WT: bf16 transposed weights [N,K]
// C:  CMODE 0: f32 row-major [M,N]
//     CMODE 1: bf16 head-major [B,NH,S,HD]
//     CMODE 2: bf16 head-transposed [B,NH,HD,S]
// 256 threads (8 waves); tile 128x64; wave computes 32x32; K-step 32.
// Double-buffered LDS fed by global_load_async_to_lds_b128 (3 per lane/step).
// ---------------------------------------------------------------------------
template<int AMODE, int CMODE>
__global__ __launch_bounds__(256) void gemm_wmma_kernel(
    const __bf16* __restrict__ Ab, const __bf16* __restrict__ WT,
    const float* __restrict__ bias,
    float* __restrict__ Cf, __bf16* __restrict__ Cb,
    int M, int N, int Kd, int S, int NH, int HD)
{
  // stride 40 bf16 = 80 B: 16B-aligned rows, conflict-free fragment reads
  __shared__ __bf16 lA[2][128][40];
  __shared__ __bf16 lB[2][64][40];

  const int t    = threadIdx.x;
  const int wid  = t >> 5;
  const int lane = t & 31;
  const int hl   = lane >> 4;
  const int ln   = lane & 15;
  const int bm   = blockIdx.y * 128;
  const int bn   = blockIdx.x * 64;
  const int wM   = (wid & 3) * 32;
  const int wN   = (wid >> 2) * 32;

  v8f acc[2][2] = {};

  auto issue_tile = [&](int k0, int buf) {
    // A tile: 128 rows x 64B = 512 chunks; 2 per lane
#pragma unroll
    for (int rnd = 0; rnd < 2; ++rnd) {
      int id  = wid * 64 + rnd * 32 + lane;
      int row = id >> 2, seg = id & 3;
      const __bf16* g;
      if (AMODE == 0) {
        g = Ab + (size_t)(bm + row) * Kd + k0 + seg * 8;
      } else {
        int rr = bm + row;
        int b = rr / S, s = rr - b * S;
        int h = k0 / HD, d = k0 - h * HD + seg * 8;
        g = Ab + (((size_t)b * NH + h) * S + s) * HD + d;
      }
      async_copy_b128(lds_off(&lA[buf][row][seg * 8]), g);
    }
    // B tile: 64 rows (n) x 64B = 256 chunks; 1 per lane
    int id = wid * 32 + lane;
    int n = id >> 2, seg = id & 3;
    async_copy_b128(lds_off(&lB[buf][n][seg * 8]),
                    WT + (size_t)(bn + n) * Kd + k0 + seg * 8);
  };

  issue_tile(0, 0);
  const int nk = Kd >> 5;
  for (int it = 0; it < nk; ++it) {
    const int buf = it & 1;
    const bool more = (it + 1) < nk;
    if (more) {
      issue_tile((it + 1) << 5, buf ^ 1);
      wait_asynccnt<3>();   // in-order completion: current tile's 3 are done
    } else {
      wait_asynccnt<0>();
    }
    __syncthreads();

    FragBF afr[2], bfr[2];
#pragma unroll
    for (int mi = 0; mi < 2; ++mi) {
      const __bf16* base = &lA[buf][wM + mi * 16 + ln][0];
#pragma unroll
      for (int vv = 0; vv < 4; ++vv) {
        afr[mi].u[vv]     = *(const unsigned*)(base + hl * 8 + 2 * vv);       // K 0..15
        afr[mi].u[4 + vv] = *(const unsigned*)(base + 16 + hl * 8 + 2 * vv);  // K 16..31
      }
    }
#pragma unroll
    for (int ni = 0; ni < 2; ++ni) {
      const __bf16* base = &lB[buf][wN + ni * 16 + ln][0];
#pragma unroll
      for (int vv = 0; vv < 8; ++vv)
        bfr[ni].u[vv] = *(const unsigned*)(base + (vv >> 2) * 16 + hl * 8 + 2 * (vv & 3));
    }
#pragma unroll
    for (int mi = 0; mi < 2; ++mi)
#pragma unroll
      for (int ni = 0; ni < 2; ++ni)
        acc[mi][ni] = wmma_bf16(afr[mi].v, bfr[ni].v, acc[mi][ni]);

    __syncthreads();   // all waves done reading buf before it is overwritten
  }

  // ---- epilogue: bias + store (C layout: lane col = ln, rows = hl*8+r) ----
#pragma unroll
  for (int mi = 0; mi < 2; ++mi) {
#pragma unroll
    for (int ni = 0; ni < 2; ++ni) {
      int col = bn + wN + ni * 16 + ln;
      float bv = bias[col];
#pragma unroll
      for (int r = 0; r < 8; ++r) {
        int row = bm + wM + mi * 16 + hl * 8 + r;
        float v = acc[mi][ni][r] + bv;
        if (CMODE == 0) {
          Cf[(size_t)row * N + col] = v;
        } else {
          int b = row / S, s = row - b * S;
          int h = col / HD, d = col - h * HD;
          if (CMODE == 1)
            Cb[(((size_t)b * NH + h) * S + s) * HD + d] = (__bf16)v;
          else
            Cb[(((size_t)b * NH + h) * HD + d) * S + s] = (__bf16)v;
        }
      }
    }
  }
}

// ---------------------------------------------------------------------------
// RoPE (interleaved-pair convention, matching the reference) on Q and K.
// ---------------------------------------------------------------------------
__global__ void rope_kernel(__bf16* __restrict__ q, __bf16* __restrict__ k,
                            int S, int HD, long long npairs)
{
  long long idx = (long long)blockIdx.x * blockDim.x + threadIdx.x;
  if (idx >= npairs) return;
  int hp = HD >> 1;
  int j = (int)(idx % hp);
  long long rest = idx / hp;          // (b*NH+h)*S + s
  int s = (int)(rest % S);
  float inv = __expf((-2.0f * (float)j / (float)HD) * 9.210340371976184f); // 10000^(-2j/HD)
  float ang = (float)s * inv;
  float sn, cs;
  __sincosf(ang, &sn, &cs);
  size_t off = (size_t)rest * HD + 2 * j;
  float qe = (float)q[off], qo = (float)q[off + 1];
  q[off]     = (__bf16)(qe * cs - qo * sn);
  q[off + 1] = (__bf16)(qo * cs + qe * sn);
  float ke = (float)k[off], ko = (float)k[off + 1];
  k[off]     = (__bf16)(ke * cs - ko * sn);
  k[off + 1] = (__bf16)(ko * cs + ke * sn);
}

// ---------------------------------------------------------------------------
// Flash attention: 8 waves x 16 query rows = 128 rows per block per (b,h).
// Streams 32-key blocks: 8 QK^T WMMAs + online softmax + 8 PV WMMAs.
// V pre-transposed per head ([B,NH,HD,S]) so PV B-frags are contiguous loads.
// ---------------------------------------------------------------------------
__global__ __launch_bounds__(256) void fattn_kernel(
    const __bf16* __restrict__ Q, const __bf16* __restrict__ K,
    const __bf16* __restrict__ VT, __bf16* __restrict__ O,
    int S, int HD, float scale)
{
  __shared__ __bf16 lP[8][16][34];   // per-wave P staging (C-layout -> A-frag)
  const int bh   = blockIdx.y;
  const int wid  = threadIdx.x >> 5;
  const int lane = threadIdx.x & 31;
  const int hl = lane >> 4, ln = lane & 15;
  const __bf16* Qh = Q  + (size_t)bh * S * HD;
  const __bf16* Kh = K  + (size_t)bh * S * HD;
  const __bf16* Vh = VT + (size_t)bh * HD * S;
  const int q0 = blockIdx.x * 128 + wid * 16;

  FragBF qf[4];
  {
    const __bf16* qrow = Qh + (size_t)(q0 + ln) * HD;
#pragma unroll
    for (int c = 0; c < 4; ++c) {
      const __bf16* base = qrow + c * 32;
#pragma unroll
      for (int vv = 0; vv < 4; ++vv) {
        qf[c].u[vv]     = *(const unsigned*)(base + hl * 8 + 2 * vv);
        qf[c].u[4 + vv] = *(const unsigned*)(base + 16 + hl * 8 + 2 * vv);
      }
    }
  }

  v8f oacc[8] = {};
  float mrun[8], lrun[8];
#pragma unroll
  for (int r = 0; r < 8; ++r) { mrun[r] = -1e30f; lrun[r] = 0.f; }

  for (int kb = 0; kb < S; kb += 32) {
    v8f s0 = {}, s1 = {};
#pragma unroll
    for (int c = 0; c < 4; ++c) {
      FragBF kf0, kf1;
      const __bf16* p0 = Kh + (size_t)(kb + ln) * HD + c * 32 + hl * 8;
      const __bf16* p1 = Kh + (size_t)(kb + 16 + ln) * HD + c * 32 + hl * 8;
#pragma unroll
      for (int vv = 0; vv < 8; ++vv) {
        kf0.u[vv] = *(const unsigned*)(p0 + (vv >> 2) * 16 + 2 * (vv & 3));
        kf1.u[vv] = *(const unsigned*)(p1 + (vv >> 2) * 16 + 2 * (vv & 3));
      }
      s0 = wmma_bf16(qf[c].v, kf0.v, s0);
      s1 = wmma_bf16(qf[c].v, kf1.v, s1);
    }

    float bmax[8];
#pragma unroll
    for (int r = 0; r < 8; ++r) {
      float a = s0[r] * scale, b = s1[r] * scale;
      s0[r] = a; s1[r] = b;
      bmax[r] = fmaxf(a, b);
    }
#pragma unroll
    for (int msk = 1; msk < 16; msk <<= 1)
#pragma unroll
      for (int r = 0; r < 8; ++r)
        bmax[r] = fmaxf(bmax[r], __shfl_xor(bmax[r], msk, 32));

    float alpha[8], rsum[8];
#pragma unroll
    for (int r = 0; r < 8; ++r) {
      float mn = fmaxf(mrun[r], bmax[r]);
      alpha[r] = __expf(mrun[r] - mn);
      mrun[r] = mn;
      float p0 = __expf(s0[r] - mn);
      float p1 = __expf(s1[r] - mn);
      s0[r] = p0; s1[r] = p1;
      rsum[r] = p0 + p1;
    }
#pragma unroll
    for (int msk = 1; msk < 16; msk <<= 1)
#pragma unroll
      for (int r = 0; r < 8; ++r)
        rsum[r] += __shfl_xor(rsum[r], msk, 32);
#pragma unroll
    for (int r = 0; r < 8; ++r)
      lrun[r] = lrun[r] * alpha[r] + rsum[r];

#pragma unroll
    for (int c = 0; c < 8; ++c)
#pragma unroll
      for (int r = 0; r < 8; ++r)
        oacc[c][r] *= alpha[r];

#pragma unroll
    for (int r = 0; r < 8; ++r) {
      lP[wid][hl * 8 + r][ln]      = (__bf16)s0[r];
      lP[wid][hl * 8 + r][16 + ln] = (__bf16)s1[r];
    }
    FragBF pf;
    {
      const __bf16* base = &lP[wid][ln][0];
#pragma unroll
      for (int vv = 0; vv < 4; ++vv) {
        pf.u[vv]     = *(const unsigned*)(base + hl * 8 + 2 * vv);
        pf.u[4 + vv] = *(const unsigned*)(base + 16 + hl * 8 + 2 * vv);
      }
    }

#pragma unroll
    for (int c = 0; c < 8; ++c) {
      FragBF vf;
      const __bf16* vb = Vh + (size_t)(c * 16 + ln) * S + kb + hl * 8;
#pragma unroll
      for (int vv = 0; vv < 8; ++vv)
        vf.u[vv] = *(const unsigned*)(vb + (vv >> 2) * 16 + 2 * (vv & 3));
      oacc[c] = wmma_bf16(pf.v, vf.v, oacc[c]);
    }
  }

#pragma unroll
  for (int r = 0; r < 8; ++r) {
    float rinv = 1.0f / lrun[r];
    size_t rowoff = (size_t)bh * S * HD + (size_t)(q0 + hl * 8 + r) * HD;
#pragma unroll
    for (int c = 0; c < 8; ++c)
      O[rowoff + c * 16 + ln] = (__bf16)(oacc[c][r] * rinv);
  }
}

// ---------------------------------------------------------------------------
extern "C" void kernel_launch(void* const* d_in, const int* in_sizes, int n_in,
                              void* d_out, int out_size, void* d_ws, size_t ws_size,
                              hipStream_t stream)
{
  const float* hidden = (const float*)d_in[0];
  const float* Wq = (const float*)d_in[1];
  const float* bq = (const float*)d_in[2];
  const float* Wk = (const float*)d_in[3];
  const float* bk = (const float*)d_in[4];
  const float* Wv = (const float*)d_in[5];
  const float* bv = (const float*)d_in[6];
  const float* Wo = (const float*)d_in[7];
  const float* bo = (const float*)d_in[8];

  int H = 1;
  while ((long long)H * H < (long long)in_sizes[1]) H <<= 1;   // H = 2048
  const int NH = 16;
  const int HD = H / NH;                  // 128
  const int B  = 2;
  const int S  = in_sizes[0] / (B * H);   // 2048
  const int M = B * S, N = H, Kd = H;

  size_t BNSH = (size_t)B * NH * (size_t)S * HD;
  size_t MK   = (size_t)M * Kd;
  size_t NK   = (size_t)N * Kd;
  __bf16* Qb  = (__bf16*)d_ws;     // [B,NH,S,HD]
  __bf16* Kb  = Qb + BNSH;         // [B,NH,S,HD]
  __bf16* Vt  = Kb + BNSH;         // [B,NH,HD,S]
  __bf16* Ob  = Vt + BNSH;         // [B,NH,S,HD]
  __bf16* Abf = Ob + BNSH;         // [M,K] bf16 hidden
  __bf16* WTq = Abf + MK;          // [N,K] bf16 transposed weights
  __bf16* WTk = WTq + NK;
  __bf16* WTv = WTk + NK;
  __bf16* WTo = WTv + NK;

  // ---- prep: bf16 conversion + weight transposes ----
  to_bf16_kernel<<<(int)((MK / 4 + 255) / 256), 256, 0, stream>>>(hidden, Abf, MK);
  dim3 tg(N / 32, Kd / 32), tb(256);
  transpose_bf16_kernel<<<tg, tb, 0, stream>>>(Wq, WTq, Kd, N);
  transpose_bf16_kernel<<<tg, tb, 0, stream>>>(Wk, WTk, Kd, N);
  transpose_bf16_kernel<<<tg, tb, 0, stream>>>(Wv, WTv, Kd, N);
  transpose_bf16_kernel<<<tg, tb, 0, stream>>>(Wo, WTo, Kd, N);

  // ---- projections (async-fed WMMA GEMMs) ----
  dim3 gg(N / 64, M / 128);
  gemm_wmma_kernel<0,1><<<gg, tb, 0, stream>>>(Abf, WTq, bq, nullptr, Qb, M, N, Kd, S, NH, HD);
  gemm_wmma_kernel<0,1><<<gg, tb, 0, stream>>>(Abf, WTk, bk, nullptr, Kb, M, N, Kd, S, NH, HD);
  gemm_wmma_kernel<0,2><<<gg, tb, 0, stream>>>(Abf, WTv, bv, nullptr, Vt, M, N, Kd, S, NH, HD);

  long long pairs = (long long)B * NH * S * (HD / 2);
  rope_kernel<<<(int)((pairs + 255) / 256), 256, 0, stream>>>(Qb, Kb, S, HD, pairs);

  float scale = 1.0f / sqrtf((float)HD);
  fattn_kernel<<<dim3(S / 128, B * NH), tb, 0, stream>>>(Qb, Kb, Vt, Ob, S, HD, scale);

  gemm_wmma_kernel<1,0><<<gg, tb, 0, stream>>>(Ob, WTo, bo, (float*)d_out, nullptr, M, N, Kd, S, NH, HD);
}
